// VectorQuantizer_31791347925309
// MI455X (gfx1250) — compile-verified, matrix operand order checked by reading, not run on hardware
//
#include <hip/hip_runtime.h>

// Vector-quantizer (VQ-VAE codebook lookup) for gfx1250 / MI455X.
//   inputs:      [64,32,32,64] f32  -> N=65536 rows of D=64
//   embeddings:  [2048,64]     f32  -> K=2048 codes
// Outputs (concatenated in d_out, all f32):
//   quantized   N*D   indices N (as float)   flat_inputs N*D
//
// argmin_k (||e_k||^2 - 2 x.e_k); x.e_k computed on the deep-K bf16 matrix
// pipe via the 3-term split  x.e ~= xh.eh + xh.el + xl.eh  (each term a
// V_WMMA_F32_16X16X32_BF16, K=32, two chunks cover D=64 -> 6 WMMA / tile).
// fp32 WMMA (16x16x4) would need 16 instrs/tile at ~1/8 the FLOP rate; the
// bf16x3 path keeps ~2^-16 relative accuracy (ample for ranking) while
// cutting matrix-pipe time ~8x. ||e||^2 stays exact fp32.

#define K_EMB 2048
#define D_DIM 64
#define N_VEC (64 * 32 * 32)        // 65536 input vectors
#define N_TILES (K_EMB / 16)        // 128 codebook column tiles
#define WAVES_PER_BLOCK 8
#define LDS_ROW 72                  // 64 bf16 + pad -> 144B row stride (16B mult.)

typedef __attribute__((ext_vector_type(4)))  float  v4f;
typedef __attribute__((ext_vector_type(8)))  float  v8f;
typedef __attribute__((ext_vector_type(4)))  __bf16 v4bf;
typedef __attribute__((ext_vector_type(8)))  __bf16 v8bf;
typedef __attribute__((ext_vector_type(16))) __bf16 v16bf;

// ---------------------------------------------------------------------------
// Pass 1: per codebook row -> ||e||^2 (exact f32) + bf16 hi/lo planes
// ---------------------------------------------------------------------------
__global__ __launch_bounds__(256) void vq_prep_kernel(const float* __restrict__ emb,
                                                      float*  __restrict__ esq,
                                                      __bf16* __restrict__ ehi,
                                                      __bf16* __restrict__ elo) {
    int k = blockIdx.x * blockDim.x + threadIdx.x;
    if (k >= K_EMB) return;
    const v4f* row = (const v4f*)(emb + (size_t)k * D_DIM);
    float s = 0.0f;
#pragma unroll
    for (int i = 0; i < D_DIM / 4; ++i) {
        v4f v = row[i];
        v4bf h, l;
#pragma unroll
        for (int j = 0; j < 4; ++j) {
            float  f  = v[j];
            __bf16 hh = (__bf16)f;                 // RNE high part
            h[j] = hh;
            l[j] = (__bf16)(f - (float)hh);        // residual
            s += f * f;
        }
        *(v4bf*)(ehi + (size_t)k * D_DIM + i * 4) = h;
        *(v4bf*)(elo + (size_t)k * D_DIM + i * 4) = l;
    }
    esq[k] = s;
}

// Glue two 8-element runs (K j..j+7 and K j+16..j+23) into a v16bf operand.
__device__ __forceinline__ v16bf combine16(const __bf16* p) {
    v8bf r0 = *(const v8bf*)p;          // elements 0..7
    v8bf r1 = *(const v8bf*)(p + 16);   // elements 8..15
    return __builtin_shufflevector(r0, r1, 0, 1, 2, 3, 4, 5, 6, 7,
                                           8, 9, 10, 11, 12, 13, 14, 15);
}

__device__ __forceinline__ v8f wmma_bf16(v16bf a, v16bf b, v8f c) {
    return __builtin_amdgcn_wmma_f32_16x16x32_bf16(
        /*neg_a=*/false, a, /*neg_b=*/false, b,
        /*c_mod=*/(short)0, c, /*reuse_a=*/false, /*reuse_b=*/false);
}

// ---------------------------------------------------------------------------
// Pass 2: bf16x3 WMMA distance scan + argmin + gather
// ---------------------------------------------------------------------------
__global__ __launch_bounds__(256) void vq_main_kernel(const float*  __restrict__ x,
                                                      const float*  __restrict__ emb,
                                                      const float*  __restrict__ esq,
                                                      const __bf16* __restrict__ ehi,
                                                      const __bf16* __restrict__ elo,
                                                      float* __restrict__ outq,
                                                      float* __restrict__ outi,
                                                      float* __restrict__ outf) {
    // ping-pong codebook {hi,lo} planes, 16 rows x 72 bf16 each (2*2*2304B = 9216B)
    __shared__ __align__(16) __bf16 ebuf[2][2][16 * LDS_ROW];

    const int tid  = threadIdx.x;
    const int wave = tid >> 5;
    const int lane = tid & 31;
    const int half = lane >> 4;     // lane group: selects K sub-runs
    const int lrow = lane & 15;     // M-row (A) / N-col (B,C) within the 16-tile
    const int mb   = (blockIdx.x * WAVES_PER_BLOCK + wave) * 16;   // M-tile base

    // --- A fragments (16-bit A 16x32 layout): lane L, element j:
    //       j<8 : K = 32*ch + 8*half + j
    //       j>=8: K = 32*ch + 16 + 8*half + (j-8)
    v16bf a_hi[2], a_lo[2];
    {
        const float* xrow = x + (size_t)(mb + lrow) * D_DIM + half * 8;
#pragma unroll
        for (int ch = 0; ch < 2; ++ch) {
#pragma unroll
            for (int j = 0; j < 8; ++j) {
                float  f0 = xrow[32 * ch + j];
                float  f1 = xrow[32 * ch + 16 + j];
                __bf16 h0 = (__bf16)f0;
                __bf16 h1 = (__bf16)f1;
                a_hi[ch][j]     = h0;
                a_hi[ch][j + 8] = h1;
                a_lo[ch][j]     = (__bf16)(f0 - (float)h0);
                a_lo[ch][j + 8] = (__bf16)(f1 - (float)h1);
            }
        }
    }

    float bd[8];      // best distance for rows m = v + 8*half
    int   bi[8];
#pragma unroll
    for (int v = 0; v < 8; ++v) { bd[v] = 3.4e38f; bi[v] = 0; }

    // Staging role: 256 threads move 2 planes x 16 rows x 8 chunks of 16B.
    const int sp   = tid >> 7;          // 0 = hi plane, 1 = lo plane
    const int srow = (tid & 127) >> 3;  // codebook row within tile
    const int schk = tid & 7;           // 16B chunk within row
    const __bf16* gsrc = (sp ? elo : ehi) + (size_t)srow * D_DIM + schk * 8;
    __bf16* ldst0 = &ebuf[0][sp][srow * LDS_ROW + schk * 8];
    __bf16* ldst1 = &ebuf[1][sp][srow * LDS_ROW + schk * 8];

    // Stage tile 0.
    *(v8bf*)ldst0 = *(const v8bf*)gsrc;
    __syncthreads();

    for (int tile = 0; tile < N_TILES; ++tile) {
        const int cur = tile & 1;

        // Prefetch next tile into the other buffer (uniform branch).
        if (tile + 1 < N_TILES) {
            *(v8bf*)(cur ? ldst0 : ldst1) =
                *(const v8bf*)(gsrc + (size_t)(tile + 1) * 16 * D_DIM);
        }

        // --- B fragments (16-bit B 32x16 layout, mirror of A): lane's col
        //     n = lrow; per chunk, two 16B runs at +32*ch and +32*ch+16.
        const __bf16* bh = &ebuf[cur][0][lrow * LDS_ROW + half * 8];
        const __bf16* bl = &ebuf[cur][1][lrow * LDS_ROW + half * 8];
        v16bf Bh0 = combine16(bh);
        v16bf Bh1 = combine16(bh + 32);
        v16bf Bl0 = combine16(bl);
        v16bf Bl1 = combine16(bl + 32);

        // bf16x3: small terms first, dominant hi*hi last.
        v8f acc = {0.f, 0.f, 0.f, 0.f, 0.f, 0.f, 0.f, 0.f};
        acc = wmma_bf16(a_hi[0], Bl0, acc);
        acc = wmma_bf16(a_hi[1], Bl1, acc);
        acc = wmma_bf16(a_lo[0], Bh0, acc);
        acc = wmma_bf16(a_lo[1], Bh1, acc);
        acc = wmma_bf16(a_hi[0], Bh0, acc);
        acc = wmma_bf16(a_hi[1], Bh1, acc);

        // distance proxy d = ||e_n||^2 - 2*(x_m . e_n); lane's n fixed per tile
        const int   n  = tile * 16 + lrow;
        const float e2 = esq[n];
#pragma unroll
        for (int v = 0; v < 8; ++v) {
            float d = e2 - 2.0f * acc[v];
            if (d < bd[v]) { bd[v] = d; bi[v] = n; }   // strict <: first idx wins
        }

        __syncthreads();   // protects ebuf ping-pong across all 8 waves
    }

    // --- argmin across the 16 lanes of each half-wave (tie -> lower index)
#pragma unroll
    for (int v = 0; v < 8; ++v) {
        float d = bd[v];
        int   i = bi[v];
#pragma unroll
        for (int mask = 8; mask >= 1; mask >>= 1) {
            float od = __shfl_xor(d, mask, 32);
            int   oi = __shfl_xor(i, mask, 32);
            if (od < d || (od == d && oi < i)) { d = od; i = oi; }
        }
        bd[v] = d;
        bi[v] = i;
    }
    // Every lane of a half now holds final results for rows m = v + 8*half.

    // --- indices output (one lane per half writes its 8 rows)
    if (lrow == 0) {
#pragma unroll
        for (int v = 0; v < 8; ++v)
            outi[mb + v + 8 * half] = (float)bi[v];
    }

    // --- gather quantized rows from the exact f32 codebook
#pragma unroll
    for (int v = 0; v < 8; ++v) {
        int row = mb + v + 8 * half;
        int idx = bi[v];
        v4f val = *(const v4f*)(emb + (size_t)idx * D_DIM + lrow * 4);
        *(v4f*)(outq + (size_t)row * D_DIM + lrow * 4) = val;
    }

    // --- flat_inputs copy: wave streams its own 16x64 tile (1024 floats)
#pragma unroll
    for (int it = 0; it < 8; ++it) {
        int off = it * 128 + lane * 4;
        *(v4f*)(outf + (size_t)mb * D_DIM + off) =
            *(const v4f*)(x + (size_t)mb * D_DIM + off);
    }
}

// ---------------------------------------------------------------------------
extern "C" void kernel_launch(void* const* d_in, const int* in_sizes, int n_in,
                              void* d_out, int out_size, void* d_ws, size_t ws_size,
                              hipStream_t stream) {
    (void)in_sizes; (void)n_in; (void)out_size; (void)ws_size;

    const float* x   = (const float*)d_in[0];   // [N, 64]
    const float* emb = (const float*)d_in[1];   // [2048, 64]

    // workspace layout: esq f32[2048] | ehi bf16[2048*64] | elo bf16[2048*64]
    float*  esq = (float*)d_ws;
    __bf16* ehi = (__bf16*)((char*)d_ws + 8192);
    __bf16* elo = (__bf16*)((char*)d_ws + 8192 + (size_t)K_EMB * D_DIM * 2);

    float* outq = (float*)d_out;                         // N*D quantized
    float* outi = outq + (size_t)N_VEC * D_DIM;          // N indices (as float)
    float* outf = outi + N_VEC;                          // N*D flat inputs

    vq_prep_kernel<<<K_EMB / 256, 256, 0, stream>>>(emb, esq, ehi, elo);

    const int tiles  = N_VEC / 16;                       // 4096 M-tiles
    const int blocks = tiles / WAVES_PER_BLOCK;          // 512 blocks of 256 thr
    vq_main_kernel<<<blocks, 256, 0, stream>>>(x, emb, esq, ehi, elo,
                                               outq, outi, outf);
}
